// QGAHybridModel_27513560498688
// MI455X (gfx1250) — matrix operand order repaired and checked by reading, not compiled, not run on hardware
//
#include <hip/hip_runtime.h>
#include <hip/hip_bf16.h>

typedef __attribute__((ext_vector_type(16))) _Float16 v16h;
typedef __attribute__((ext_vector_type(8)))  float    v8f;

#define G_    64
#define O_    256
#define DIN_  128
#define H_    64
#define NODES 16384          // G_*O_
#define E_    524288         // G_*8192
#define BN_EPS 1e-5f

// ---------------- WMMA fragment loaders (wave32, 16x16x32 f16) ----------------
// A (16x32 f16): elem j of lane L -> M = mrow + (L&15), K = kbase + (L>>4)*16 + j
__device__ __forceinline__ v16h ldsA_frag(const _Float16* sA, int ldk, int mrow, int kbase, int lane) {
  const _Float16* p = sA + (mrow + (lane & 15)) * ldk + kbase + ((lane >> 4) << 4);
  v16h a;
#pragma unroll
  for (int j = 0; j < 16; ++j) a[j] = p[j];
  return a;
}
// B (32x16 f16) staged n-major/k-minor: elem j of lane L -> N = nbase + (L&15), K = kbase + (L>>4)*16 + j
__device__ __forceinline__ v16h ldsBt_frag(const _Float16* sBt, int ldk, int nbase, int kbase, int lane) {
  const _Float16* p = sBt + (nbase + (lane & 15)) * ldk + kbase + ((lane >> 4) << 4);
  v16h b;
#pragma unroll
  for (int j = 0; j < 16; ++j) b[j] = p[j];
  return b;
}

// ---------------- omics encoder: per-modality [64x128]@[128x64] + rot/ReLU/BN ----------------
__global__ __launch_bounds__(128)
void omics_encoder(const float* __restrict__ omics, const float* __restrict__ w_omics,
                   const float* __restrict__ b_omics, const float* __restrict__ rot,
                   const float* __restrict__ gam, const float* __restrict__ bet,
                   const float* __restrict__ mean, const float* __restrict__ var,
                   float* __restrict__ X0) {
  __shared__ _Float16 sA[G_ * DIN_];    // [g][d]
  __shared__ _Float16 sBt[H_ * DIN_];   // [h][d]
  const int o = blockIdx.x, t = threadIdx.x;
  for (int idx = t; idx < G_ * DIN_; idx += 128) {
    int g = idx >> 7, d = idx & 127;
    sA[idx] = (_Float16)omics[((long)g * O_ + o) * DIN_ + d];
  }
  const float* W = w_omics + (long)o * DIN_ * H_;
  for (int idx = t; idx < DIN_ * H_; idx += 128) {
    int d = idx >> 6, h = idx & 63;
    sBt[h * DIN_ + d] = (_Float16)W[idx];
  }
  __syncthreads();
  const int lane = t & 31, w = t >> 5;
  v8f acc[4];
#pragma unroll
  for (int i = 0; i < 4; ++i)
#pragma unroll
    for (int r = 0; r < 8; ++r) acc[i][r] = 0.f;
#pragma unroll
  for (int ks = 0; ks < 4; ++ks) {
    v16h a = ldsA_frag(sA, DIN_, 16 * w, ks * 32, lane);
#pragma unroll
    for (int nt = 0; nt < 4; ++nt) {
      v16h b = ldsBt_frag(sBt, DIN_, nt * 16, ks * 32, lane);
      acc[nt] = __builtin_amdgcn_wmma_f32_16x16x32_f16(false, a, false, b, (short)0, acc[nt], false, false);
    }
  }
  const int col = lane & 15, rsel = (lane >> 4) << 3;
#pragma unroll
  for (int nt = 0; nt < 4; ++nt) {
    int h = nt * 16 + col;
    float bo = b_omics[o * H_ + h];
    float rv = rot[o * H_ + h];
    float rf = __cosf(rv) + __sinf(rv);
    float inv = rsqrtf(var[o * H_ + h] + BN_EPS);
    float ga = gam[o * H_ + h], be = bet[o * H_ + h], mu = mean[o * H_ + h];
#pragma unroll
    for (int r = 0; r < 8; ++r) {
      int g = 16 * w + rsel + r;
      float v = (acc[nt][r] + bo) * rf;
      v = v > 0.f ? v : 0.f;                 // ReLU
      v = (v - mu) * inv * ga + be;          // BN eval
      X0[((long)g * O_ + o) * H_ + h] = v;
    }
  }
}

// ---------------- generic node GEMM: C[16384xN] = A[16384xK] @ B[KxN] ----------------
template<int K, int N>
__global__ __launch_bounds__(128)
void gemm_node(const float* __restrict__ A, const float* __restrict__ B, float* __restrict__ C) {
  __shared__ _Float16 sA[64 * K];
  __shared__ _Float16 sBt[N * 32];
  const int t = threadIdx.x;
  const long row0 = (long)blockIdx.x * 64;
  for (int idx = t; idx < 64 * K; idx += 128) sA[idx] = (_Float16)A[row0 * K + idx];
  const int lane = t & 31, w = t >> 5;
  v8f acc[N / 16];
#pragma unroll
  for (int i = 0; i < N / 16; ++i)
#pragma unroll
    for (int r = 0; r < 8; ++r) acc[i][r] = 0.f;
#pragma unroll 1
  for (int ks = 0; ks < K / 32; ++ks) {
    __syncthreads();
    if (ks + 1 < K / 32) __builtin_prefetch(B + (ks + 1) * 32 * N + t, 0, 0);
    for (int idx = t; idx < 32 * N; idx += 128) {
      int kk = idx / N, n = idx % N;
      sBt[n * 32 + kk] = (_Float16)B[(ks * 32 + kk) * N + n];
    }
    __syncthreads();
    v16h a = ldsA_frag(sA, K, 16 * w, ks * 32, lane);
#pragma unroll
    for (int nt = 0; nt < N / 16; ++nt) {
      v16h b = ldsBt_frag(sBt, 32, nt * 16, 0, lane);
      acc[nt] = __builtin_amdgcn_wmma_f32_16x16x32_f16(false, a, false, b, (short)0, acc[nt], false, false);
    }
  }
  const int col = lane & 15, rbase = 16 * w + ((lane >> 4) << 3);
#pragma unroll
  for (int nt = 0; nt < N / 16; ++nt)
#pragma unroll
    for (int r = 0; r < 8; ++r)
      C[(row0 + rbase + r) * N + nt * 16 + col] = acc[nt][r];
}

// ---------------- attention helpers ----------------
__device__ __forceinline__ unsigned ord_enc(float f) {
  unsigned u = __float_as_uint(f);
  return (u & 0x80000000u) ? ~u : (u | 0x80000000u);
}
__device__ __forceinline__ float ord_dec(unsigned u) {
  u = (u & 0x80000000u) ? (u & 0x7fffffffu) : ~u;
  return __uint_as_float(u);
}

__global__ void fill_u32(unsigned* p, unsigned v, int n) {
  int i = blockIdx.x * blockDim.x + threadIdx.x;
  if (i < n) p[i] = v;
}

// per-node a_src/a_dst dot products; blockDim = heads*64
__global__ void attn_coef(const float* __restrict__ Hbuf, const float* __restrict__ av_s,
                          const float* __restrict__ av_d, float* __restrict__ AS,
                          float* __restrict__ AD, int heads) {
  __shared__ float ss[256], sd[256];
  const int node = blockIdx.x, t = threadIdx.x;
  float hv = Hbuf[(long)node * blockDim.x + t];
  ss[t] = hv * av_s[t];
  sd[t] = hv * av_d[t];
  __syncthreads();
  for (int off = 32; off > 0; off >>= 1) {
    if ((t & 63) < off) { ss[t] += ss[t + off]; sd[t] += sd[t + off]; }
    __syncthreads();
  }
  if ((t & 63) == 0) {
    int hd = t >> 6;
    AS[node * heads + hd] = ss[t];
    AD[node * heads + hd] = sd[t];
  }
}

__global__ void edge_max_k(const int* __restrict__ srcv, const int* __restrict__ dstv,
                           const float* __restrict__ AS, const float* __restrict__ AD,
                           unsigned* __restrict__ MAXI, int total, int heads) {
  int i = blockIdx.x * blockDim.x + threadIdx.x;
  if (i >= total) return;
  int e = i / heads, hd = i - e * heads;
  float l = AS[srcv[e] * heads + hd] + AD[dstv[e] * heads + hd];
  l = l > 0.f ? l : 0.2f * l;                     // leaky_relu 0.2
  atomicMax(&MAXI[dstv[e] * heads + hd], ord_enc(l));
}

__global__ void edge_sum_k(const int* __restrict__ srcv, const int* __restrict__ dstv,
                           const float* __restrict__ AS, const float* __restrict__ AD,
                           const unsigned* __restrict__ MAXI, float* __restrict__ SUM,
                           int total, int heads) {
  int i = blockIdx.x * blockDim.x + threadIdx.x;
  if (i >= total) return;
  int e = i / heads, hd = i - e * heads;
  int d = dstv[e];
  float l = AS[srcv[e] * heads + hd] + AD[d * heads + hd];
  l = l > 0.f ? l : 0.2f * l;
  float m = ord_dec(MAXI[d * heads + hd]);
  if (!(m > -3.0e38f)) m = 0.f;                   // isfinite fix
  atomicAdd(&SUM[d * heads + hd], __expf(l - m));
}

template<int HEADS>
__global__ void edge_msg_k(const int* __restrict__ srcv, const int* __restrict__ dstv,
                           const float* __restrict__ AS, const float* __restrict__ AD,
                           const unsigned* __restrict__ MAXI, const float* __restrict__ SUM,
                           const float* __restrict__ Hbuf, float* __restrict__ OUT, int E) {
  const int C = HEADS * 64;
  const int epb = 256 / C;
  int t = threadIdx.x;
  int e = blockIdx.x * epb + t / C;
  if (e >= E) return;
  int lt = t % C, hd = lt >> 6;
  int s = srcv[e], d = dstv[e];
  float l = AS[s * HEADS + hd] + AD[d * HEADS + hd];
  l = l > 0.f ? l : 0.2f * l;
  float m = ord_dec(MAXI[d * HEADS + hd]);
  if (!(m > -3.0e38f)) m = 0.f;
  float alpha = __expf(l - m) / (SUM[d * HEADS + hd] + 1e-16f);
  atomicAdd(&OUT[(long)d * C + lt], Hbuf[(long)s * C + lt] * alpha);
}

__global__ void elu_bias(float* __restrict__ X, const float* __restrict__ bias, int n, int cmask) {
  int i = blockIdx.x * blockDim.x + threadIdx.x;
  if (i < n) {
    float v = X[i] + bias[i & cmask];
    X[i] = v > 0.f ? v : (__expf(v) - 1.f);
  }
}
__global__ void add_bias(float* __restrict__ X, const float* __restrict__ bias, int n, int cmask) {
  int i = blockIdx.x * blockDim.x + threadIdx.x;
  if (i < n) X[i] += bias[i & cmask];
}

__global__ void pool_k(const float* __restrict__ X2, float* __restrict__ pooled) {
  int g = blockIdx.x, h = threadIdx.x;
  float s = 0.f;
  for (int n = 0; n < O_; ++n) s += X2[((long)g * O_ + n) * H_ + h];
  pooled[g * H_ + h] = s * (1.f / O_);
}

__global__ void cls_k(const float* __restrict__ pooled, const float* __restrict__ w1,
                      const float* __restrict__ b1, const float* __restrict__ w2,
                      const float* __restrict__ b2, float* __restrict__ out) {
  __shared__ float sr[64];
  int g = blockIdx.x, h = threadIdx.x;
  float a = b1[h];
  for (int c = 0; c < 64; ++c) a += pooled[g * 64 + c] * w1[c * 64 + h];
  a = a > 0.f ? a : 0.f;
  sr[h] = a * w2[h];
  __syncthreads();
  for (int off = 32; off > 0; off >>= 1) {
    if (h < off) sr[h] += sr[h + off];
    __syncthreads();
  }
  if (h == 0) out[g] = 1.f / (1.f + __expf(-(sr[0] + b2[0])));
}

extern "C" void kernel_launch(void* const* d_in, const int* in_sizes, int n_in,
                              void* d_out, int out_size, void* d_ws, size_t ws_size,
                              hipStream_t stream) {
  const float* omics   = (const float*)d_in[0];
  const float* w_omics = (const float*)d_in[1];
  const float* b_omics = (const float*)d_in[2];
  const float* rot     = (const float*)d_in[3];
  const float* bn_g    = (const float*)d_in[4];
  const float* bn_b    = (const float*)d_in[5];
  const float* bn_m    = (const float*)d_in[6];
  const float* bn_v    = (const float*)d_in[7];
  const float* gat1_w  = (const float*)d_in[8];
  const float* as1v    = (const float*)d_in[9];
  const float* ad1v    = (const float*)d_in[10];
  const float* bias1   = (const float*)d_in[11];
  const float* gat2_w  = (const float*)d_in[12];
  const float* as2v    = (const float*)d_in[13];
  const float* ad2v    = (const float*)d_in[14];
  const float* bias2   = (const float*)d_in[15];
  const float* cls_w1  = (const float*)d_in[16];
  const float* cls_b1  = (const float*)d_in[17];
  const float* cls_w2  = (const float*)d_in[18];
  const float* cls_b2  = (const float*)d_in[19];
  const int*   eidx    = (const int*)d_in[20];
  const int*   srcv    = eidx;
  const int*   dstv    = eidx + E_;

  float* ws = (float*)d_ws;
  size_t off = 0;
  float* X0  = ws + off;  off += (size_t)NODES * H_;
  float* H1  = ws + off;  off += (size_t)NODES * 256;
  float* AS1 = ws + off;  off += (size_t)NODES * 4;
  float* AD1 = ws + off;  off += (size_t)NODES * 4;
  unsigned* MAX1 = (unsigned*)(ws + off); off += (size_t)NODES * 4;
  float* SUM1 = ws + off; off += (size_t)NODES * 4;
  float* X1  = ws + off;  off += (size_t)NODES * 256;
  float* H2  = ws + off;  off += (size_t)NODES * H_;
  float* AS2 = ws + off;  off += (size_t)NODES;
  float* AD2 = ws + off;  off += (size_t)NODES;
  unsigned* MAX2 = (unsigned*)(ws + off); off += (size_t)NODES;
  float* SUM2 = ws + off; off += (size_t)NODES;
  float* X2  = ws + off;  off += (size_t)NODES * H_;

  float* outp   = (float*)d_out;        // [0,64): sigmoid out
  float* pooled = outp + G_;            // [64,4160): pooled

  auto cdiv = [](int a, int b) { return (a + b - 1) / b; };
  const unsigned NEG_INF_ENC = 0x007fffffu;   // ord_enc(-inf)

  // 1) omics encoder -> X0 [NODES x 64]
  omics_encoder<<<O_, 128, 0, stream>>>(omics, w_omics, b_omics, rot, bn_g, bn_b, bn_m, bn_v, X0);
  // 2) GAT1 linear -> H1 [NODES x 256]
  gemm_node<64, 256><<<NODES / 64, 128, 0, stream>>>(X0, gat1_w, H1);
  // 3) attention coefficients (heads=4)
  attn_coef<<<NODES, 256, 0, stream>>>(H1, as1v, ad1v, AS1, AD1, 4);
  // 4) init segment buffers
  fill_u32<<<cdiv(NODES * 4, 256), 256, 0, stream>>>(MAX1, NEG_INF_ENC, NODES * 4);
  fill_u32<<<cdiv(NODES * 4, 256), 256, 0, stream>>>((unsigned*)SUM1, 0u, NODES * 4);
  fill_u32<<<cdiv(NODES * 256, 256), 256, 0, stream>>>((unsigned*)X1, 0u, NODES * 256);
  // 5) segment max / exp-sum / weighted scatter
  edge_max_k<<<cdiv(E_ * 4, 256), 256, 0, stream>>>(srcv, dstv, AS1, AD1, MAX1, E_ * 4, 4);
  edge_sum_k<<<cdiv(E_ * 4, 256), 256, 0, stream>>>(srcv, dstv, AS1, AD1, MAX1, SUM1, E_ * 4, 4);
  edge_msg_k<4><<<E_, 256, 0, stream>>>(srcv, dstv, AS1, AD1, MAX1, SUM1, H1, X1, E_);
  // 6) X1 = elu(X1 + bias1)
  elu_bias<<<cdiv(NODES * 256, 256), 256, 0, stream>>>(X1, bias1, NODES * 256, 255);
  // 7) GAT2 linear -> H2 [NODES x 64]
  gemm_node<256, 64><<<NODES / 64, 128, 0, stream>>>(X1, gat2_w, H2);
  // 8) attention (heads=1)
  attn_coef<<<NODES, 64, 0, stream>>>(H2, as2v, ad2v, AS2, AD2, 1);
  fill_u32<<<cdiv(NODES, 256), 256, 0, stream>>>(MAX2, NEG_INF_ENC, NODES);
  fill_u32<<<cdiv(NODES, 256), 256, 0, stream>>>((unsigned*)SUM2, 0u, NODES);
  fill_u32<<<cdiv(NODES * H_, 256), 256, 0, stream>>>((unsigned*)X2, 0u, NODES * H_);
  edge_max_k<<<cdiv(E_, 256), 256, 0, stream>>>(srcv, dstv, AS2, AD2, MAX2, E_, 1);
  edge_sum_k<<<cdiv(E_, 256), 256, 0, stream>>>(srcv, dstv, AS2, AD2, MAX2, SUM2, E_, 1);
  edge_msg_k<1><<<E_ / 4, 256, 0, stream>>>(srcv, dstv, AS2, AD2, MAX2, SUM2, H2, X2, E_);
  // 9) bias2, mean-pool, classifier
  add_bias<<<cdiv(NODES * H_, 256), 256, 0, stream>>>(X2, bias2, NODES * H_, 63);
  pool_k<<<G_, H_, 0, stream>>>(X2, pooled);
  cls_k<<<G_, 64, 0, stream>>>(pooled, cls_w1, cls_b1, cls_w2, cls_b2, outp);

  (void)in_sizes; (void)n_in; (void)out_size; (void)ws_size; (void)d_in;
}
